// SparseMultiHeadAttention_19576460935442
// MI455X (gfx1250) — matrix operand order repaired and checked by reading, not compiled, not run on hardware
//
#include <hip/hip_runtime.h>
#include <hip/hip_bf16.h>

typedef __attribute__((ext_vector_type(16))) __bf16 v16bf;
typedef __attribute__((ext_vector_type(8)))  __bf16 v8bf;
typedef __attribute__((ext_vector_type(8)))  float  v8f;

#define N_NODES 50000
#define N_EDGES 800000
#define DIM     128
#define NHEAD   8
#define HDIM    16
#define LDS_STRIDE 136   // halves; 272 B row stride: 16B-aligned + conflict-spreading
#define WPACK_ELEMS (4 * DIM * DIM)

// Packed-B fragment layout: fragment f = (t*4 + kb) with t = m*8 + c.
// 512 bf16 per fragment = 32 lanes x 16 contiguous elements (32 B/lane).
// Element i of lane L = W[kb*32 + i + (i>=8?8:0) + (L>=16?8:0)][c*16 + (L&15)]
// per the ISA 16-bit B 32x16 VGPR layout. Tail threads pack biases into bcat.
__global__ void pack_weights(const float* __restrict__ Wq, const float* __restrict__ Wk,
                             const float* __restrict__ Wv, const float* __restrict__ Wo,
                             const float* __restrict__ bq, const float* __restrict__ bk,
                             const float* __restrict__ bv, const float* __restrict__ bo,
                             __bf16* __restrict__ wpack, float* __restrict__ bcat) {
    int t = blockIdx.x * 256 + threadIdx.x;
    if (t < WPACK_ELEMS) {
        int i    = t & 15;
        int lane = (t >> 4) & 31;
        int kb   = (t >> 9) & 3;
        int c    = (t >> 11) & 7;
        int m    = t >> 14;
        const float* src = (m == 0) ? Wq : (m == 1) ? Wk : (m == 2) ? Wv : Wo;
        int K    = kb * 32 + i + ((i >> 3) << 3) + (lane >> 4) * 8;
        int ncol = c * 16 + (lane & 15);
        wpack[t] = (__bf16)src[(size_t)K * DIM + ncol];
    } else if (t < WPACK_ELEMS + 4 * DIM) {
        int idx = t - WPACK_ELEMS;
        int m = idx >> 7, j = idx & 127;
        const float* bsrc = (m == 0) ? bq : (m == 1) ? bk : (m == 2) ? bv : bo;
        bcat[idx] = bsrc[j];
    }
}

__global__ void init_segmax(float* __restrict__ segmax) {
    int t = blockIdx.x * 256 + threadIdx.x;
    if (t < N_NODES * NHEAD) segmax[t] = -3.0e38f;
}

// Build v16bf A operand from LDS: two 16B contiguous runs per lane.
__device__ __forceinline__ v16bf load_A(const __bf16* sx, int mrow, int k0, int aoff) {
    const __bf16* ap = sx + mrow * LDS_STRIDE + k0 + aoff;
    v8bf a0 = *(const v8bf*)(ap);        // A elements 0..7  : K = k0+aoff+0..7
    v8bf a1 = *(const v8bf*)(ap + 16);   // A elements 8..15 : K = k0+aoff+16..23
    return __builtin_shufflevector(a0, a1, 0, 1, 2, 3, 4, 5, 6, 7,
                                           8, 9, 10, 11, 12, 13, 14, 15);
}

// ---------------- fused QKV projection with WMMA bf16 ----------------
// grid.x = N/16 row tiles, 128 threads = 4 waves. 24 output tiles (3 matrices
// x 8 col tiles); each wave does 3 iterations of an independent tile PAIR so
// interleaved WMMAs fill each other's hazard slots.
__global__ __launch_bounds__(128) void qkv_proj(
    const float* __restrict__ x, const __bf16* __restrict__ wpack,
    const float* __restrict__ bcat, float* __restrict__ qkv) {
    __shared__ __bf16 sx[16 * LDS_STRIDE];
    const int tid = threadIdx.x;
    const int row0 = blockIdx.x * 16;
    for (int i = tid; i < 16 * DIM; i += 128) {
        int r = i >> 7, c = i & 127;
        sx[r * LDS_STRIDE + c] = (__bf16)x[(size_t)(row0 + r) * DIM + c];
    }
    __syncthreads();

    const int wave = tid >> 5;
    const int lane = tid & 31;
    const int mrow = lane & 15;
    const int aoff = (lane >> 4) * 8;   // A K-offset / C row-offset for upper half-wave

    #pragma unroll
    for (int tt = 0; tt < 3; ++tt) {
        const int t0 = wave + tt * 8;   // pair (t0, t0+4): independent accumulators
        const int t1 = t0 + 4;
        const __bf16* wp0 = wpack + (size_t)(t0 * 4) * 512 + lane * 16;
        const __bf16* wp1 = wpack + (size_t)(t1 * 4) * 512 + lane * 16;
        v8f acc0 = {}, acc1 = {};
        #pragma unroll
        for (int kb = 0; kb < 4; ++kb) {
            v16bf A  = load_A(sx, mrow, kb * 32, aoff);
            v16bf B0 = *(const v16bf*)(wp0 + kb * 512);
            v16bf B1 = *(const v16bf*)(wp1 + kb * 512);
            acc0 = __builtin_amdgcn_wmma_f32_16x16x32_bf16(
                false, A, false, B0, (short)0, acc0, false, false);
            acc1 = __builtin_amdgcn_wmma_f32_16x16x32_bf16(
                false, A, false, B1, (short)0, acc1, false, false);
        }
        // t = m*8+c  =>  bias index = m*128 + c*16 + mrow = t*16 + mrow
        const float bc0 = bcat[t0 * 16 + mrow];
        const float bc1 = bcat[t1 * 16 + mrow];
        float* o0 = qkv + (size_t)(t0 >> 3) * N_NODES * DIM + ((t0 & 7) * 16 + mrow);
        float* o1 = qkv + (size_t)(t1 >> 3) * N_NODES * DIM + ((t1 & 7) * 16 + mrow);
        #pragma unroll
        for (int r = 0; r < 8; ++r) {
            int mr = r + aoff;          // C/D layout: upper half-wave holds rows M+8
            o0[(size_t)(row0 + mr) * DIM] = acc0[r] + bc0;
            o1[(size_t)(row0 + mr) * DIM] = acc1[r] + bc1;
        }
    }
}

// ---------------- per-(edge,head) attention logits ----------------
__global__ void edge_logits(const int* __restrict__ ei, const float* __restrict__ ebias,
                            const float* __restrict__ q, const float* __restrict__ k,
                            float* __restrict__ attn) {
    int t = blockIdx.x * 256 + threadIdx.x;
    if (t >= N_EDGES * NHEAD) return;
    int e = t >> 3, h = t & 7;
    int row = ei[e];
    int col = ei[N_EDGES + e];
    const float4* qv = (const float4*)(q + (size_t)col * DIM + h * HDIM);
    const float4* kv = (const float4*)(k + (size_t)row * DIM + h * HDIM);
    float s = 0.f;
    #pragma unroll
    for (int i = 0; i < 4; ++i) {
        float4 a = qv[i], b = kv[i];
        s += a.x * b.x + a.y * b.y + a.z * b.z + a.w * b.w;
    }
    attn[t] = s * 0.25f + ebias[t];   // scale = HD^-0.5 = 0.25
}

// ---------------- scatter max over destination node ----------------
__global__ void seg_max_kernel(const int* __restrict__ ei, const float* __restrict__ attn,
                               float* __restrict__ segmax) {
    int t = blockIdx.x * 256 + threadIdx.x;
    if (t >= N_EDGES * NHEAD) return;
    int e = t >> 3, h = t & 7;
    int col = ei[N_EDGES + e];
    float val = attn[t];
    unsigned int* ua = (unsigned int*)(segmax + (size_t)col * NHEAD + h);
    unsigned int old = *ua;
    while (__uint_as_float(old) < val) {
        unsigned int assumed = old;
        old = atomicCAS(ua, assumed, __float_as_uint(val));
        if (old == assumed) break;
    }
}

// ---------------- exp + denominator ----------------
__global__ void edge_exp(const int* __restrict__ ei, float* __restrict__ attn,
                         const float* __restrict__ segmax, float* __restrict__ denom) {
    int t = blockIdx.x * 256 + threadIdx.x;
    if (t >= N_EDGES * NHEAD) return;
    int e = t >> 3, h = t & 7;
    int col = ei[N_EDGES + e];
    float ex = __expf(attn[t] - segmax[(size_t)col * NHEAD + h]);
    attn[t] = ex;
    atomicAdd(&denom[(size_t)col * NHEAD + h], ex);
}

// ---------------- weighted scatter of V ----------------
__global__ void edge_scatter(const int* __restrict__ ei, const float* __restrict__ attn,
                             const float* __restrict__ denom, const float* __restrict__ v,
                             float* __restrict__ acc) {
    long long t = (long long)blockIdx.x * 256 + threadIdx.x;
    if (t >= (long long)N_EDGES * DIM) return;
    int e = (int)(t >> 7);
    int d = (int)(t & 127);
    int h = d >> 4;
    int row = ei[e];
    int col = ei[N_EDGES + e];
    float w = attn[(size_t)e * NHEAD + h] / (denom[(size_t)col * NHEAD + h] + 1e-16f);
    atomicAdd(&acc[(size_t)col * DIM + d], w * v[(size_t)row * DIM + d]);
}

// ---------------- output projection acc @ Wo + bo with WMMA ----------------
__global__ __launch_bounds__(128) void out_proj(
    const float* __restrict__ acc, const __bf16* __restrict__ wpack,
    const float* __restrict__ bcat, float* __restrict__ out) {
    __shared__ __bf16 sx[16 * LDS_STRIDE];
    const int tid = threadIdx.x;
    const int row0 = blockIdx.x * 16;
    for (int i = tid; i < 16 * DIM; i += 128) {
        int r = i >> 7, c = i & 127;
        sx[r * LDS_STRIDE + c] = (__bf16)acc[(size_t)(row0 + r) * DIM + c];
    }
    __syncthreads();

    const int wave = tid >> 5;
    const int lane = tid & 31;
    const int mrow = lane & 15;
    const int aoff = (lane >> 4) * 8;

    // two column tiles per wave (c and c+4), interleaved accumulators
    const int c0 = wave, c1 = wave + 4;
    const int ncol0 = c0 * 16 + mrow, ncol1 = c1 * 16 + mrow;
    const __bf16* wp0 = wpack + (size_t)((24 + c0) * 4) * 512 + lane * 16;  // Wo = m3
    const __bf16* wp1 = wpack + (size_t)((24 + c1) * 4) * 512 + lane * 16;
    v8f acc0 = {}, acc1 = {};
    #pragma unroll
    for (int kb = 0; kb < 4; ++kb) {
        v16bf A  = load_A(sx, mrow, kb * 32, aoff);
        v16bf B0 = *(const v16bf*)(wp0 + kb * 512);
        v16bf B1 = *(const v16bf*)(wp1 + kb * 512);
        acc0 = __builtin_amdgcn_wmma_f32_16x16x32_bf16(
            false, A, false, B0, (short)0, acc0, false, false);
        acc1 = __builtin_amdgcn_wmma_f32_16x16x32_bf16(
            false, A, false, B1, (short)0, acc1, false, false);
    }
    const float bc0 = bcat[3 * DIM + ncol0];
    const float bc1 = bcat[3 * DIM + ncol1];
    #pragma unroll
    for (int r = 0; r < 8; ++r) {
        int mr = r + aoff;
        out[(size_t)(row0 + mr) * DIM + ncol0] = acc0[r] + bc0;
        out[(size_t)(row0 + mr) * DIM + ncol1] = acc1[r] + bc1;
    }
}

extern "C" void kernel_launch(void* const* d_in, const int* in_sizes, int n_in,
                              void* d_out, int out_size, void* d_ws, size_t ws_size,
                              hipStream_t stream) {
    const float* x     = (const float*)d_in[0];
    const int*   ei    = (const int*)  d_in[1];   // [2, E]: row=ei[0:E], col=ei[E:2E]
    const float* ebias = (const float*)d_in[2];
    const float* Wq    = (const float*)d_in[3];
    const float* bq    = (const float*)d_in[4];
    const float* Wk    = (const float*)d_in[5];
    const float* bk    = (const float*)d_in[6];
    const float* Wv    = (const float*)d_in[7];
    const float* bv    = (const float*)d_in[8];
    const float* Wo    = (const float*)d_in[9];
    const float* bo    = (const float*)d_in[10];
    float* out = (float*)d_out;

    // workspace layout (floats); wpack lands 32B-aligned after the f32 arrays
    float* qkv    = (float*)d_ws;                        // q|k|v contiguous
    float* q      = qkv;
    float* k      = q   + (size_t)N_NODES * DIM;
    float* v      = k   + (size_t)N_NODES * DIM;
    float* acc    = v   + (size_t)N_NODES * DIM;
    float* denom  = acc + (size_t)N_NODES * DIM;         // adjacent to acc: one memset
    float* attn   = denom + (size_t)N_NODES * NHEAD;
    float* segmax = attn  + (size_t)N_EDGES * NHEAD;
    __bf16* wpack = (__bf16*)(segmax + (size_t)N_NODES * NHEAD);
    float*  bcat  = (float*)(wpack + WPACK_ELEMS);       // 4*128 f32 biases

    // 1) weights -> bf16 pre-packed into per-lane WMMA B fragments (+ bias concat)
    pack_weights<<<(WPACK_ELEMS + 4 * DIM + 255) / 256, 256, 0, stream>>>(
        Wq, Wk, Wv, Wo, bq, bk, bv, bo, wpack, bcat);
    // 2) init reductions
    init_segmax<<<(N_NODES * NHEAD + 255) / 256, 256, 0, stream>>>(segmax);
    hipMemsetAsync(acc, 0, (size_t)(N_NODES * DIM + N_NODES * NHEAD) * sizeof(float), stream);
    // 3) q,k,v projections (WMMA)
    qkv_proj<<<N_NODES / 16, 128, 0, stream>>>(x, wpack, bcat, qkv);
    // 4) edge logits
    edge_logits<<<(N_EDGES * NHEAD) / 256, 256, 0, stream>>>(ei, ebias, q, k, attn);
    // 5) scatter softmax
    seg_max_kernel<<<(N_EDGES * NHEAD) / 256, 256, 0, stream>>>(ei, attn, segmax);
    edge_exp<<<(N_EDGES * NHEAD) / 256, 256, 0, stream>>>(ei, attn, segmax, denom);
    // 6) weighted scatter of V
    edge_scatter<<<(unsigned)(((long long)N_EDGES * DIM) / 256), 256, 0, stream>>>(ei, attn, denom, v, acc);
    // 7) output projection (WMMA)
    out_proj<<<N_NODES / 16, 128, 0, stream>>>(acc, wpack, bcat, out);
}
